// HierarchicalMultiHeadCrossAttention_58669253263601
// MI455X (gfx1250) — matrix operand order repaired and checked by reading, not compiled, not run on hardware
//
#include <hip/hip_runtime.h>
#include <hip/hip_bf16.h>

typedef __bf16 bf16_t;
typedef __attribute__((ext_vector_type(16))) __bf16 v16bf;
typedef __attribute__((ext_vector_type(8)))  float  v8f;

#define DIM        512
#define NUM_HEADS  8
#define HEAD_DIM   64
#define KV_LEN     2048
#define NUM_LEVELS 11
#define NUM_NODES  2047
#define EPSV       1e-7f

// ---------------------------------------------------------------------------
// WMMA helper: D = A(16x32 bf16) * B(32x16 bf16) + C(16x16 f32)
// ---------------------------------------------------------------------------
__device__ inline v8f wmma_bf16(v16bf a, v16bf b, v8f c) {
  return __builtin_amdgcn_wmma_f32_16x16x32_bf16(
      /*neg_a=*/false, a, /*neg_b=*/false, b,
      /*c_mod=*/(short)0, c, /*reuse_a=*/false, /*reuse_b=*/false);
}

// A fragment (16x32) from f32 row-major, converting to bf16.
// m = lane&15; lanes 0-15 hold K 0-7 & 16-23, lanes 16-31 hold K 8-15 & 24-31.
__device__ inline v16bf frag_a_f32(const float* A, int lda, int row0, int k0, int lane) {
  const int m  = lane & 15;
  const int kh = (lane >> 4) * 8;
  const float* p = A + (size_t)(row0 + m) * lda + k0 + kh;
  v16bf a;
#pragma unroll
  for (int e = 0; e < 8; ++e) a[e] = (__bf16)p[e];
#pragma unroll
  for (int e = 0; e < 8; ++e) a[8 + e] = (__bf16)p[16 + e];
  return a;
}

__device__ inline v16bf frag_a_bf16(const bf16_t* A, int lda, int row0, int k0, int lane) {
  const int m  = lane & 15;
  const int kh = (lane >> 4) * 8;
  const bf16_t* p = A + (size_t)(row0 + m) * lda + k0 + kh;
  v16bf a;
#pragma unroll
  for (int e = 0; e < 8; ++e) a[e] = p[e];
#pragma unroll
  for (int e = 0; e < 8; ++e) a[8 + e] = p[16 + e];
  return a;
}

// B fragment from pre-packed bf16: pack[((kt*Npad)+n)*32 + kw], so each lane
// reads 2 x b128. n = col0+(lane&15), kw = (lane>>4)*8 + {0-7, 16-23}.
__device__ inline v16bf frag_b_packed(const bf16_t* pack, int Npad, int kt,
                                      int col0, int lane) {
  const int n  = col0 + (lane & 15);
  const int kh = (lane >> 4) * 8;
  const bf16_t* p = pack + ((size_t)kt * Npad + n) * 32 + kh;
  v16bf b;
#pragma unroll
  for (int e = 0; e < 8; ++e) b[e] = p[e];
#pragma unroll
  for (int e = 0; e < 8; ++e) b[8 + e] = p[16 + e];
  return b;
}

// ---------------------------------------------------------------------------
// P0: pack f32 W[K,N] -> bf16 fragment-linear layout, zero-padded to Npad cols
// ---------------------------------------------------------------------------
__global__ void __launch_bounds__(256)
pack_w(const float* __restrict__ W, bf16_t* __restrict__ P, int K, int N, int Npad) {
  const int o = blockIdx.x * 256 + threadIdx.x;
  const int total = (K >> 5) * Npad * 32;
  if (o >= total) return;
  const int kw = o & 31;
  const int n  = (o >> 5) % Npad;
  const int kt = (o >> 5) / Npad;
  const int k  = kt * 32 + kw;
  const float v = (n < N) ? W[(size_t)k * N + n] : 0.0f;
  P[o] = (__bf16)v;
}

// ---------------------------------------------------------------------------
// K1: q = X[4096,512] @ Wq + bq -> bf16 q_flat, head-permuted row-major
//     qf[((b*8+h)*1024 + s)*64 + dc].  One 16x64 tile per wave.
// ---------------------------------------------------------------------------
__global__ void __launch_bounds__(256)
proj_gemm_q(const float* __restrict__ X, const bf16_t* __restrict__ WP,
            const float* __restrict__ bias, bf16_t* __restrict__ dst) {
  const int lane = threadIdx.x & 31;
  const int wave = threadIdx.x >> 5;
  const int task = blockIdx.x * 8 + wave;  // 2048 tasks
  const int row0 = (task & 255) << 4;      // mtiles = 256
  const int col0 = (task >> 8) << 6;       // 8 n-groups of 64

  v8f acc[4] = {};
#pragma unroll 1
  for (int k0 = 0; k0 < DIM; k0 += 32) {
    const int kt = k0 >> 5;
    const v16bf a = frag_a_f32(X, DIM, row0, k0, lane);
#pragma unroll
    for (int nt = 0; nt < 4; ++nt)
      acc[nt] = wmma_bf16(a, frag_b_packed(WP, DIM, kt, col0 + nt * 16, lane), acc[nt]);
  }
  const int n  = lane & 15;
  const int rh = (lane >> 4) * 8;
#pragma unroll
  for (int nt = 0; nt < 4; ++nt) {
    const int col = col0 + nt * 16 + n;
    const float bv = bias[col];
    const int h  = col >> 6;
    const int dc = col & 63;
#pragma unroll
    for (int i = 0; i < 8; ++i) {
      const int r  = row0 + rh + i;
      const int bb = r >> 10;
      const int s  = r & 1023;
      dst[((size_t)(bb * NUM_HEADS + h) * 1024 + s) * HEAD_DIM + dc] =
          (__bf16)(acc[nt][i] + bv);
    }
  }
}

// ---------------------------------------------------------------------------
// K2: v = X[8192,512] @ Wv + bv -> bf16, stored in packed-B layout per (b,h):
//     vpack[bh][kt=kv>>5][dc][kv&31]. One 16x64 tile per wave.
// ---------------------------------------------------------------------------
__global__ void __launch_bounds__(256)
proj_gemm_v(const float* __restrict__ X, const bf16_t* __restrict__ WP,
            const float* __restrict__ bias, bf16_t* __restrict__ dst) {
  const int lane = threadIdx.x & 31;
  const int wave = threadIdx.x >> 5;
  const int task = blockIdx.x * 8 + wave;  // 4096 tasks
  const int row0 = (task & 511) << 4;      // mtiles = 512
  const int col0 = (task >> 9) << 6;

  v8f acc[4] = {};
#pragma unroll 1
  for (int k0 = 0; k0 < DIM; k0 += 32) {
    const int kt = k0 >> 5;
    const v16bf a = frag_a_f32(X, DIM, row0, k0, lane);
#pragma unroll
    for (int nt = 0; nt < 4; ++nt)
      acc[nt] = wmma_bf16(a, frag_b_packed(WP, DIM, kt, col0 + nt * 16, lane), acc[nt]);
  }
  const int n  = lane & 15;
  const int rh = (lane >> 4) * 8;
#pragma unroll
  for (int nt = 0; nt < 4; ++nt) {
    const int col = col0 + nt * 16 + n;
    const float bv = bias[col];
    const int h  = col >> 6;
    const int dc = col & 63;
#pragma unroll
    for (int i = 0; i < 8; ++i) {
      const int r  = row0 + rh + i;
      const int bb = r >> 11;
      const int s  = r & 2047;  // kv index
      const size_t idx =
          ((((size_t)(bb * NUM_HEADS + h) * 64 + (s >> 5)) * 64 + dc) << 5) + (s & 31);
      dst[idx] = (__bf16)(acc[nt][i] + bv);
    }
  }
}

// ---------------------------------------------------------------------------
// K3: fused routing GEMM + sigmoid + binary-tree expansion + mask/normalize
//     + attn@V for a tile of 16 query rows of one (b,h).
// ---------------------------------------------------------------------------
#define SMEM_D_BYTES    (16 * 2048 * 4)  // d:    16 x 2048 f32  (128 KB)
#define SMEM_ATTN_BYTES (16 * 2048 * 2)  // attn: 16 x 2048 bf16 ( 64 KB)

__global__ void __launch_bounds__(256)
fused_route_attn(const bf16_t* __restrict__ qf, const bf16_t* __restrict__ vpack,
                 const bf16_t* __restrict__ WrP, const float* __restrict__ br,
                 const int* __restrict__ mask, bf16_t* __restrict__ ctx) {
  __shared__ __align__(16) unsigned char smem[SMEM_D_BYTES + SMEM_ATTN_BYTES + 2112];
  float*  d_s     = (float*)smem;                                    // [16][2048]
  bf16_t* attn_s  = (bf16_t*)(smem + SMEM_D_BYTES);                  // [16][2048]
  float*  s1a     = (float*)(smem + SMEM_D_BYTES + SMEM_ATTN_BYTES); // [16][16]
  float*  s2a     = s1a + 256;                                       // [16][16]
  float*  scale_s = s2a + 256;                                       // [16]
  float*  ctxp    = d_s;  // reused after phase 2: [8][16][64] partial ctx

  const int lane = threadIdx.x & 31;
  const int wave = threadIdx.x >> 5;
  const int bh   = blockIdx.x >> 6;      // b*8 + h
  const int qt   = blockIdx.x & 63;
  const int b    = bh >> 3;
  const int row0 = bh * 1024 + qt * 16;  // row in q_flat [32768, 64]

  // ---- Phase 1: d = sigmoid(q_tile @ Wr + br) -> LDS ----------------------
  const v16bf a0 = frag_a_bf16(qf, HEAD_DIM, row0, 0,  lane);
  const v16bf a1 = frag_a_bf16(qf, HEAD_DIM, row0, 32, lane);
#pragma unroll 1
  for (int t = wave * 16; t < wave * 16 + 16; ++t) {  // 128 node-tiles total
    const int col0 = t << 4;
    v8f acc = {};
    acc = wmma_bf16(a0, frag_b_packed(WrP, 2048, 0, col0, lane), acc);
    acc = wmma_bf16(a1, frag_b_packed(WrP, 2048, 1, col0, lane), acc);
    const int node = col0 + (lane & 15);
    if (node < NUM_NODES) {
      const float bb = br[node];
#pragma unroll
      for (int i = 0; i < 8; ++i) {
        const int r = (lane >> 4) * 8 + i;
        const float x = acc[i] + bb;
        d_s[r * 2048 + node] = 1.0f / (1.0f + __expf(-x));
      }
    }
  }
  __syncthreads();

  // ---- Phase 2: binary-tree leaf products, mask, row sums -----------------
  {
    const int r   = threadIdx.x >> 4;
    const int sub = threadIdx.x & 15;
    const float* drow = d_s + r * 2048;
    const int*   mrow = mask + b * KV_LEN;
    float s1 = 0.0f, s2 = 0.0f;
#pragma unroll 1
    for (int jj = 0; jj < 128; ++jj) {
      const int j = jj * 16 + sub;
      float p = 1.0f;
#pragma unroll
      for (int lvl = 0; lvl < NUM_LEVELS; ++lvl) {
        const int node = ((1 << lvl) - 1) + (j >> (NUM_LEVELS - lvl));
        const float dv = drow[node];
        p *= ((j >> (NUM_LEVELS - 1 - lvl)) & 1) ? (1.0f - dv) : dv;
      }
      s1 += p;
      const float pm = p * (float)mrow[j];
      s2 += pm;
      attn_s[r * 2048 + j] = (__bf16)pm;
    }
    s1a[r * 16 + sub] = s1;
    s2a[r * 16 + sub] = s2;
  }
  __syncthreads();
  if (threadIdx.x < 16) {
    float S1 = 0.0f, S2 = 0.0f;
#pragma unroll
    for (int i = 0; i < 16; ++i) {
      S1 += s1a[threadIdx.x * 16 + i];
      S2 += s2a[threadIdx.x * 16 + i];
    }
    // leaf/(S1+eps) -> mask -> /(S2'+eps) collapses to one per-row scale
    scale_s[threadIdx.x] = 1.0f / (S2 + EPSV * (S1 + EPSV));
  }
  __syncthreads();

  // ---- Phase 3: ctx[16,64] = attn[16,2048] @ v[2048,64], K split 8 ways ---
  {
    const bf16_t* vb = vpack + (size_t)bh * KV_LEN * HEAD_DIM;  // packed per bh
    v8f acc[4] = {};
    const int kbase = wave * 256;
#pragma unroll 1
    for (int ks = 0; ks < 256; ks += 32) {
      const int k0 = kbase + ks;
      const v16bf a = frag_a_bf16(attn_s, 2048, 0, k0, lane);
#pragma unroll
      for (int nt = 0; nt < 4; ++nt)
        acc[nt] = wmma_bf16(a, frag_b_packed(vb, 64, k0 >> 5, nt * 16, lane), acc[nt]);
    }
    const int n = lane & 15;
#pragma unroll
    for (int i = 0; i < 8; ++i) {
      const int rr = (lane >> 4) * 8 + i;
      float* dst = ctxp + (wave * 16 + rr) * 64;
#pragma unroll
      for (int nt = 0; nt < 4; ++nt) dst[nt * 16 + n] = acc[nt][i];
    }
  }
  __syncthreads();

  // cross-wave reduce, apply row scale, store ctx tile (bf16, row-major)
#pragma unroll
  for (int e = 0; e < 4; ++e) {
    const int o  = threadIdx.x * 4 + e;  // 0..1023 over 16x64
    const int rr = o >> 6;
    const int c  = o & 63;
    float s = 0.0f;
#pragma unroll
    for (int w = 0; w < 8; ++w) s += ctxp[w * 1024 + o];
    s *= scale_s[rr];
    const int gr = b * 1024 + qt * 16 + rr;
    const int h  = bh & 7;
    ctx[(size_t)gr * DIM + h * HEAD_DIM + c] = (__bf16)s;
  }
}

// ---------------------------------------------------------------------------
// K4: out = ctx[4096,512](bf16) @ Wo + bo -> f32. One 16x64 tile per wave.
// ---------------------------------------------------------------------------
__global__ void __launch_bounds__(256)
out_gemm(const bf16_t* __restrict__ A, const bf16_t* __restrict__ WP,
         const float* __restrict__ bias, float* __restrict__ out) {
  const int lane = threadIdx.x & 31;
  const int wave = threadIdx.x >> 5;
  const int task = blockIdx.x * 8 + wave;  // 2048 tasks
  const int row0 = (task & 255) << 4;
  const int col0 = (task >> 8) << 6;

  v8f acc[4] = {};
#pragma unroll 1
  for (int k0 = 0; k0 < DIM; k0 += 32) {
    const int kt = k0 >> 5;
    const v16bf a = frag_a_bf16(A, DIM, row0, k0, lane);
#pragma unroll
    for (int nt = 0; nt < 4; ++nt)
      acc[nt] = wmma_bf16(a, frag_b_packed(WP, DIM, kt, col0 + nt * 16, lane), acc[nt]);
  }
  const int n  = lane & 15;
  const int rh = (lane >> 4) * 8;
#pragma unroll
  for (int nt = 0; nt < 4; ++nt) {
    const int col = col0 + nt * 16 + n;
    const float bv = bias[col];
#pragma unroll
    for (int i = 0; i < 8; ++i)
      out[(size_t)(row0 + rh + i) * DIM + col] = acc[nt][i] + bv;
  }
}

// ---------------------------------------------------------------------------
extern "C" void kernel_launch(void* const* d_in, const int* in_sizes, int n_in,
                              void* d_out, int out_size, void* d_ws, size_t ws_size,
                              hipStream_t stream) {
  const float* query = (const float*)d_in[0];   // [4,1024,512]
  const float* kvin  = (const float*)d_in[1];   // [4,2048,512]
  const int*   mask  = (const int*)d_in[2];     // [4,2048]
  const float* Wq    = (const float*)d_in[3];
  const float* bq    = (const float*)d_in[4];
  const float* Wv    = (const float*)d_in[5];
  const float* bv    = (const float*)d_in[6];
  const float* Wr    = (const float*)d_in[7];   // [64,2047]
  const float* br    = (const float*)d_in[8];
  const float* Wo    = (const float*)d_in[9];
  const float* bo    = (const float*)d_in[10];
  float* out = (float*)d_out;

  char* ws = (char*)d_ws;
  size_t off = 0;
  bf16_t* qf   = (bf16_t*)(ws + off); off += (size_t)32768 * 64 * 2;        // 4 MB
  bf16_t* vpk  = (bf16_t*)(ws + off); off += (size_t)32 * 2048 * 64 * 2;    // 8 MB
  bf16_t* ctx  = (bf16_t*)(ws + off); off += (size_t)4096 * 512 * 2;        // 4 MB
  bf16_t* WqP  = (bf16_t*)(ws + off); off += (size_t)512 * 512 * 2;         // 512 KB
  bf16_t* WvP  = (bf16_t*)(ws + off); off += (size_t)512 * 512 * 2;
  bf16_t* WoP  = (bf16_t*)(ws + off); off += (size_t)512 * 512 * 2;
  bf16_t* WrP  = (bf16_t*)(ws + off); off += (size_t)2 * 2048 * 32 * 2;     // 256 KB

  // P0: pack weights to fragment-linear bf16 (one-time per launch)
  pack_w<<<1024, 256, 0, stream>>>(Wq, WqP, 512, 512, 512);
  pack_w<<<1024, 256, 0, stream>>>(Wv, WvP, 512, 512, 512);
  pack_w<<<1024, 256, 0, stream>>>(Wo, WoP, 512, 512, 512);
  pack_w<<< 512, 256, 0, stream>>>(Wr, WrP, 64, 2047, 2048);

  // K1: q projection -> head-permuted bf16 q_flat
  proj_gemm_q<<<256, 256, 0, stream>>>(query, WqP, bq, qf);
  // K2: v projection -> packed-B bf16 v
  proj_gemm_v<<<512, 256, 0, stream>>>(kvin, WvP, bv, vpk);
  // K3: fused routing + tree expansion + normalize + attn@V
  fused_route_attn<<<2048, 256, 0, stream>>>(qf, vpk, WrP, br, mask, ctx);
  // K4: output projection -> d_out
  out_gemm<<<256, 256, 0, stream>>>(ctx, WoP, bo, out);
}
